// ManualBlock_24000277250330
// MI455X (gfx1250) — matrix operand order repaired and checked by reading, not compiled, not run on hardware
//
#include <hip/hip_runtime.h>

typedef __attribute__((ext_vector_type(16))) _Float16 v16h;
typedef __attribute__((ext_vector_type(8)))  _Float16 v8h;
typedef __attribute__((ext_vector_type(2)))  _Float16 v2h;
typedef __attribute__((ext_vector_type(8)))  float    v8f;
typedef __attribute__((ext_vector_type(4)))  float    f32x4;

#define HD   16
#define SQ   2048
#define DIM  128
#define WIN  512
#define QT   16            // queries per wave
#define KT   32            // keys per tile
#define WAVES 4
#define QB   (QT * WAVES)  // 64 queries per block
#define VT_STRIDE 40       // padded key-stride for transposed V (16B aligned rows)
#define NEGBIG (-1e9f)

__global__ __launch_bounds__(128)
void swa_attn_fwd(const float* __restrict__ Q, const float* __restrict__ K,
                  const float* __restrict__ V, float* __restrict__ O) {
  __shared__ _Float16 sK[2][KT][DIM];        // K tile, row-major [key][dim]
  __shared__ _Float16 sV[2][DIM][VT_STRIDE]; // V tile, transposed [dim][key]
  __shared__ _Float16 sP[WAVES][QT][KT];     // per-wave P relayout buffer

  const int tid  = threadIdx.x;
  const int wave = tid >> 5;
  const int lane = tid & 31;
  const int ln   = lane & 15;   // column / row-in-half selector
  const int hh   = lane >> 4;   // which 16-lane half

  const int nqb    = SQ / QB;            // 32 q-blocks per head
  const int h      = blockIdx.x / nqb;
  const int qblk   = blockIdx.x % nqb;
  const int q_base = qblk * QB;
  const int q0     = q_base + wave * QT; // this wave's first query row

  const float* Qh = Q + (size_t)h * SQ * DIM;
  const float* Kh = K + (size_t)h * SQ * DIM;
  const float* Vh = V + (size_t)h * SQ * DIM;
  float*       Oh = O + (size_t)h * SQ * DIM;

  // ---- preload Q in 16-bit A-matrix layout (16x32 per K-chunk), f32->f16 ----
  v16h qa[4];
  {
    const float* qrow = Qh + (size_t)(q0 + ln) * DIM;
#pragma unroll
    for (int kk = 0; kk < 4; ++kk) {
      const int base = kk * 32 + hh * 16;
      f32x4 a0 = *(const f32x4*)(qrow + base);
      f32x4 a1 = *(const f32x4*)(qrow + base + 4);
      f32x4 b0 = *(const f32x4*)(qrow + base + 8);
      f32x4 b1 = *(const f32x4*)(qrow + base + 12);
      v16h qv;
#pragma unroll
      for (int i = 0; i < 4; ++i) {
        qv[i]      = (_Float16)a0[i];
        qv[4 + i]  = (_Float16)a1[i];
        qv[8 + i]  = (_Float16)b0[i];
        qv[12 + i] = (_Float16)b1[i];
      }
      qa[kk] = qv;
    }
  }

  // all-ones B operand: P @ ones gives row sums of P in every lane (16x16 C layout)
  v16h onesv;
#pragma unroll
  for (int i = 0; i < 16; ++i) onesv[i] = (_Float16)1.0f;

  // ---- accumulators / softmax state ----
  v8f acc[8];
#pragma unroll
  for (int nd = 0; nd < 8; ++nd) {
    v8f z = {};
    acc[nd] = z;
  }
  v8f l_acc = {};          // row sums of P, accumulated by WMMA
  float m_i[8];
#pragma unroll
  for (int r = 0; r < 8; ++r) m_i[r] = -1e30f;

  // ---- key-tile range (union over the block's 64 queries) ----
  int klo = q_base - (WIN - 1);
  if (klo < 0) klo = 0;
  const int t_first = klo / KT;
  const int t_last  = (q_base + QB - 1) / KT;

  // cooperative stage of one 32-key tile: K row-major f16, V transposed f16
  auto stage = [&](int t, int b) {
    const int k0 = t * KT;
#pragma unroll
    for (int i = 0; i < 8; ++i) {
      const int e   = i * 512 + tid * 4;  // 4096 f32 elements per matrix
      const int row = e >> 7;             // key within tile
      const int col = e & 127;            // dim
      f32x4 dk = *(const f32x4*)(Kh + (size_t)(k0 + row) * DIM + col);
      _Float16* pk = &sK[b][row][col];
      pk[0] = (_Float16)dk.x; pk[1] = (_Float16)dk.y;
      pk[2] = (_Float16)dk.z; pk[3] = (_Float16)dk.w;
      f32x4 dv = *(const f32x4*)(Vh + (size_t)(k0 + row) * DIM + col);
      sV[b][col + 0][row] = (_Float16)dv.x;
      sV[b][col + 1][row] = (_Float16)dv.y;
      sV[b][col + 2][row] = (_Float16)dv.z;
      sV[b][col + 3][row] = (_Float16)dv.w;
    }
  };

  stage(t_first, 0);
  __syncthreads();

  for (int t = t_first; t <= t_last; ++t) {
    const int buf = t & 1;
    // prefetch tile t+2 into L2 while t+1 stages and t computes
    if (t + 2 <= t_last) {
      const float* pk = Kh + (size_t)(t + 2) * KT * DIM;
      const float* pv = Vh + (size_t)(t + 2) * KT * DIM;
      __builtin_prefetch(pk + tid * 32, 0, 0);
      __builtin_prefetch(pv + tid * 32, 0, 0);
    }
    if (t < t_last) stage(t + 1, buf ^ 1);

    const int k0 = t * KT;
    // wave-uniform: does this tile intersect this wave's window?
    if (k0 <= q0 + QT - 1 && k0 + KT - 1 >= q0 - (WIN - 1)) {
      // ---- S = Q @ K^T : two 16-key column tiles, K-dim in 4 chunks of 32 ----
      v8f c[2];
#pragma unroll
      for (int nt = 0; nt < 2; ++nt) {
        v8f cc = {};
#pragma unroll
        for (int kk = 0; kk < 4; ++kk) {
          const v16h bk = *(const v16h*)&sK[buf][nt * 16 + ln][kk * 32 + hh * 16];
          cc = __builtin_amdgcn_wmma_f32_16x16x32_f16(
              false, qa[kk], false, bk, (short)0, cc, false, false);
        }
        c[nt] = cc;
      }

      // ---- masking: only boundary tiles need it (wave-uniform branch) ----
      const bool need_mask =
          !((k0 + KT - 1 <= q0) && (k0 > q0 + QT - 1 - WIN));
      if (need_mask) {
#pragma unroll
        for (int r = 0; r < 8; ++r) {
          const int q_idx = q0 + r + hh * 8;
          const int kA = k0 + ln;
          const int kB = k0 + 16 + ln;
          const bool okA = (kA <= q_idx) && (kA > q_idx - WIN);
          const bool okB = (kB <= q_idx) && (kB > q_idx - WIN);
          c[0][r] = okA ? c[0][r] : NEGBIG;
          c[1][r] = okB ? c[1][r] : NEGBIG;
        }
      }

      // ---- row max, packed f16 butterfly (softmax is invariant to m precision,
      //      it only must keep exp(c-m) <= ~1 for the f16 P conversion) ----
      v2h pm[4];
#pragma unroll
      for (int j = 0; j < 4; ++j) {
        const float v0 = fmaxf(c[0][2 * j],     c[1][2 * j]);
        const float v1 = fmaxf(c[0][2 * j + 1], c[1][2 * j + 1]);
        v2h p;
        p[0] = (_Float16)v0;
        p[1] = (_Float16)v1;
        pm[j] = p;
      }
#pragma unroll
      for (int mask = 8; mask >= 1; mask >>= 1) {
#pragma unroll
        for (int j = 0; j < 4; ++j) {
          const int xi = __shfl_xor(__builtin_bit_cast(int, pm[j]), mask, 16);
          const v2h other = __builtin_bit_cast(v2h, xi);
          pm[j] = __builtin_elementwise_max(pm[j], other);
        }
      }

      // ---- online softmax update (no sum shuffles: row sum comes from WMMA) ----
      float scale[8];
#pragma unroll
      for (int r = 0; r < 8; ++r) {
        const float tmax  = (float)pm[r >> 1][r & 1];
        const float m_new = fmaxf(m_i[r], tmax);
        scale[r] = __expf(m_i[r] - m_new);
        m_i[r] = m_new;
        c[0][r] = __expf(c[0][r] - m_new);
        c[1][r] = __expf(c[1][r] - m_new);
      }
#pragma unroll
      for (int nd = 0; nd < 8; ++nd)
#pragma unroll
        for (int r = 0; r < 8; ++r) acc[nd][r] *= scale[r];
#pragma unroll
      for (int r = 0; r < 8; ++r) l_acc[r] *= scale[r];

      // ---- relayout P (C layout -> 16-bit A layout) through per-wave LDS ----
#pragma unroll
      for (int r = 0; r < 8; ++r) {
        sP[wave][r + hh * 8][ln]      = (_Float16)c[0][r];
        sP[wave][r + hh * 8][16 + ln] = (_Float16)c[1][r];
      }
      asm volatile("s_wait_dscnt 0x0" ::: "memory");
      v16h pa;
      {
        v8h p0 = *(const v8h*)&sP[wave][ln][hh * 16];
        v8h p1 = *(const v8h*)&sP[wave][ln][hh * 16 + 8];
#pragma unroll
        for (int i = 0; i < 8; ++i) { pa[i] = p0[i]; pa[8 + i] = p1[i]; }
      }

      // ---- l += P @ ones : row-sum via the matrix pipe ----
      l_acc = __builtin_amdgcn_wmma_f32_16x16x32_f16(
          false, pa, false, onesv, (short)0, l_acc, false, false);

      // ---- O += P @ V : 8 dim-tiles of 16, K = 32 keys ----
#pragma unroll
      for (int nd = 0; nd < 8; ++nd) {
        const v16h bv = *(const v16h*)&sV[buf][nd * 16 + ln][hh * 16];
        acc[nd] = __builtin_amdgcn_wmma_f32_16x16x32_f16(
            false, pa, false, bv, (short)0, acc[nd], false, false);
      }
    }
    __syncthreads();
  }

  // ---- normalize and store ----
  float inv[8];
#pragma unroll
  for (int r = 0; r < 8; ++r) inv[r] = 1.0f / l_acc[r];
#pragma unroll
  for (int nd = 0; nd < 8; ++nd) {
#pragma unroll
    for (int r = 0; r < 8; ++r) {
      const int row = q0 + r + hh * 8;
      const int col = nd * 16 + ln;
      Oh[(size_t)row * DIM + col] = acc[nd][r] * inv[r];
    }
  }
}

extern "C" void kernel_launch(void* const* d_in, const int* in_sizes, int n_in,
                              void* d_out, int out_size, void* d_ws, size_t ws_size,
                              hipStream_t stream) {
  const float* q = (const float*)d_in[0];
  const float* k = (const float*)d_in[1];
  const float* v = (const float*)d_in[2];
  // d_in[3] is the additive mask; it is reproduced analytically in-kernel.
  float* o = (float*)d_out;
  dim3 grid(HD * (SQ / QB));  // 16 heads * 32 query blocks = 512
  dim3 block(128);            // 4 waves of 32
  hipLaunchKernelGGL(swa_attn_fwd, grid, block, 0, stream, q, k, v, o);
}
